// CrossAttention_52982716563979
// MI455X (gfx1250) — matrix operand order repaired and checked by reading, not compiled, not run on hardware
//
#include <hip/hip_runtime.h>
#include <hip/hip_bf16.h>
#include <math.h>

// ---- problem constants (from reference) ----
#define NB1 48
#define NB2 48
#define RR  36
#define LL  30
#define DD  1024
#define NH  4
#define INV_SQRT_D 0.03125f   // 1/sqrt(1024)
#define EPSN 1e-8f

typedef __attribute__((ext_vector_type(16))) __bf16 v16bf;
typedef __attribute__((ext_vector_type(8)))  __bf16 v8bf;
typedef __attribute__((ext_vector_type(8)))  float  v8f;

#define SHUF16(lo,hi) __builtin_shufflevector(lo,hi,0,1,2,3,4,5,6,7,8,9,10,11,12,13,14,15)

__device__ __forceinline__ v8f wmma_bf16(v16bf a, v16bf b, v8f c) {
    // D = A(16x32 bf16) x B(32x16 bf16) + C(f32)
    return __builtin_amdgcn_wmma_f32_16x16x32_bf16(
        /*neg_a=*/false, a, /*neg_b=*/false, b,
        /*c_mod=*/(short)0, c, /*reuse_a=*/false, /*reuse_b=*/false);
}

// ---------------------------------------------------------------------------
// Kernel 0: f32 -> bf16 conversion
// ---------------------------------------------------------------------------
__global__ void cvt_f32_bf16(const float* __restrict__ s, __bf16* __restrict__ d, int n) {
    int i = blockIdx.x * blockDim.x + threadIdx.x;
    if (i < n) d[i] = (__bf16)s[i];
}

// ---------------------------------------------------------------------------
// Kernel 1: K = X(Mx1024) @ W(1024x1024, row = out-feature)^T, bf16 in/out,
// f32 accumulate.  One wave per 16x16 output tile.  M is a multiple of 16.
// ---------------------------------------------------------------------------
__global__ __launch_bounds__(256) void gemm_xw_kernel(
    const __bf16* __restrict__ X, const __bf16* __restrict__ W,
    __bf16* __restrict__ K, int Mrows)
{
    int wavesPerBlk = blockDim.x >> 5;
    int wid  = blockIdx.x * wavesPerBlk + (threadIdx.x >> 5);
    int mtiles = Mrows >> 4;
    int total  = mtiles * (DD >> 4);
    if (wid >= total) return;

    int mt = wid >> 6;         // 64 N-tiles
    int nt = wid & 63;
    int lane = threadIdx.x & 31;
    int m    = lane & 15;      // row (A) / col (B) index within tile
    int half = lane >> 4;

    const __bf16* arow = X + (size_t)(mt * 16 + m) * DD;            // A: row m
    const __bf16* brow = W + (size_t)(nt * 16 + m) * DD + half * 16; // B: col m

    v8f acc = {};
#pragma unroll 4
    for (int ks = 0; ks < DD; ks += 32) {
        v8bf alo = *(const v8bf*)(arow + ks + half * 16);
        v8bf ahi = *(const v8bf*)(arow + ks + half * 16 + 8);
        v16bf A = SHUF16(alo, ahi);
        v16bf B = *(const v16bf*)(brow + ks);
        acc = wmma_bf16(A, B, acc);
    }
#pragma unroll
    for (int i = 0; i < 8; ++i)
        K[(size_t)(mt * 16 + i + 8 * half) * DD + nt * 16 + m] = (__bf16)acc[i];
}

// ---------------------------------------------------------------------------
// Kernel 2: fully fused per-(a,b) pipeline -> score[a,b]
// LDS layout (bytes):
//   [0,12288)      region0: S (32x48 f32) early; later sA/sB/qA/qB/red/coA/coB
//   [12288,16384)  p1  bf16 32x64 (zero padded)
//   [16384,19456)  p2  bf16 48x32 (zero padded)
//   [19456,93184)  xA  bf16 36x1024  (fused_v1)
//   [93184,154624) xB  bf16 30x1024  (fused_v2)
// ---------------------------------------------------------------------------
#define SM_BYTES 154624
#define OFF_P1   12288
#define OFF_P2   16384
#define OFF_XA   19456
#define OFF_XB   93184

__global__ __launch_bounds__(256) void fused_score_kernel(
    const __bf16* __restrict__ k1h, const __bf16* __restrict__ k2h,
    const __bf16* __restrict__ v1h, const __bf16* __restrict__ v2h,
    const __bf16* __restrict__ w1ah, const __bf16* __restrict__ w1bh,
    const float* __restrict__ b1a, const float* __restrict__ b1b,
    const float* __restrict__ w2a, const float* __restrict__ w2b,
    float* __restrict__ out)
{
    __shared__ __align__(16) char smem[SM_BYTES];

    const int a = blockIdx.y, b = blockIdx.x;
    const int tid  = threadIdx.x;
    const int lane = tid & 31;
    const int wid  = tid >> 5;
    const int m    = lane & 15;
    const int half = lane >> 4;

    __bf16* p1 = (__bf16*)(smem + OFF_P1);
    __bf16* p2 = (__bf16*)(smem + OFF_P2);
    __bf16* xA = (__bf16*)(smem + OFF_XA);
    __bf16* xB = (__bf16*)(smem + OFF_XB);

    // zero p1/p2 pads (7168 B = 1792 dwords)
    {
        unsigned* z = (unsigned*)(smem + OFF_P1);
        for (int i = tid; i < 1792; i += 256) z[i] = 0u;
    }

    // ---- Phase 1: S[l][r] = (k2[b] . k1[a]^T) / sqrt(D)  (LDS f32, 32x48) ----
    if (wid < 6) {
        int mt = wid / 3, nt = wid % 3;
        int l = mt * 16 + m;        bool av = l < LL;
        int r = nt * 16 + m;        bool bv = r < RR;
        const __bf16* arow = k2h + (size_t)(b * LL + (av ? l : 0)) * DD;
        const __bf16* brow = k1h + (size_t)(a * RR + (bv ? r : 0)) * DD + half * 16;
        v8bf z8 = {};
        v16bf Bz = {};
        v8f acc = {};
#pragma unroll 4
        for (int ks = 0; ks < DD; ks += 32) {
            v8bf alo = av ? *(const v8bf*)(arow + ks + half * 16)     : z8;
            v8bf ahi = av ? *(const v8bf*)(arow + ks + half * 16 + 8) : z8;
            v16bf A = SHUF16(alo, ahi);
            v16bf B = bv ? *(const v16bf*)(brow + ks) : Bz;
            acc = wmma_bf16(A, B, acc);
        }
        float* S = (float*)smem;
#pragma unroll
        for (int i = 0; i < 8; ++i)
            S[(mt * 16 + i + 8 * half) * 48 + nt * 16 + m] = acc[i] * INV_SQRT_D;
    }
    __syncthreads();

    // ---- Phase 2: softmaxes -> p1 (L x R, over r), p2 (R x L, over l) ----
    {
        const float* S = (const float*)smem;
        if (tid < LL) {                     // p1 row softmax
            int l = tid;
            float mx = -1e30f;
            for (int r = 0; r < RR; ++r) mx = fmaxf(mx, S[l * 48 + r]);
            float sum = 0.f;
            for (int r = 0; r < RR; ++r) sum += __expf(S[l * 48 + r] - mx);
            float inv = 1.f / sum;
            for (int r = 0; r < RR; ++r)
                p1[l * 64 + r] = (__bf16)(__expf(S[l * 48 + r] - mx) * inv);
        }
        if (tid >= 64 && tid < 64 + RR) {   // p2 column softmax
            int r = tid - 64;
            float mx = -1e30f;
            for (int l = 0; l < LL; ++l) mx = fmaxf(mx, S[l * 48 + r]);
            float sum = 0.f;
            for (int l = 0; l < LL; ++l) sum += __expf(S[l * 48 + r] - mx);
            float inv = 1.f / sum;
            for (int l = 0; l < LL; ++l)
                p2[r * 32 + l] = (__bf16)(__expf(S[l * 48 + r] - mx) * inv);
        }
    }
    __syncthreads();

    // ---- Phase 3: fused_v2 = p1 (30x36) @ v1[a] (36x1024) -> xB bf16 ----
    for (int t = wid; t < 128; t += 8) {
        int mt = t >> 6, nt = t & 63;
        const __bf16* prow = p1 + (mt * 16 + m) * 64 + half * 16;
        int col = nt * 16 + m;
        v8f acc = {};
#pragma unroll
        for (int ks = 0; ks < 64; ks += 32) {
            v8bf alo = *(const v8bf*)(prow + ks);
            v8bf ahi = *(const v8bf*)(prow + ks + 8);
            v16bf A = SHUF16(alo, ahi);
            v16bf B = {};
#pragma unroll
            for (int e = 0; e < 16; ++e) {
                int k = ks + half * 16 + e;
                B[e] = (k < RR) ? v1h[(size_t)(a * RR + k) * DD + col] : (__bf16)0.f;
            }
            acc = wmma_bf16(A, B, acc);
        }
#pragma unroll
        for (int i = 0; i < 8; ++i) {
            int row = mt * 16 + i + 8 * half;
            if (row < LL) xB[(size_t)row * DD + col] = (__bf16)acc[i];
        }
    }
    // ---- Phase 4: fused_v1 = p2 (36x30) @ v2[b] (30x1024) -> xA bf16 ----
    for (int t = wid; t < 192; t += 8) {
        int mt = t >> 6, nt = t & 63;
        const __bf16* prow = p2 + (mt * 16 + m) * 32 + half * 16;
        int col = nt * 16 + m;
        v8bf alo = *(const v8bf*)(prow);
        v8bf ahi = *(const v8bf*)(prow + 8);
        v16bf A = SHUF16(alo, ahi);
        v16bf B = {};
#pragma unroll
        for (int e = 0; e < 16; ++e) {
            int k = half * 16 + e;
            B[e] = (k < LL) ? v2h[(size_t)(b * LL + k) * DD + col] : (__bf16)0.f;
        }
        v8f acc = {};
        acc = wmma_bf16(A, B, acc);
#pragma unroll
        for (int i = 0; i < 8; ++i) {
            int row = mt * 16 + i + 8 * half;
            if (row < RR) xA[(size_t)row * DD + col] = (__bf16)acc[i];
        }
    }
    __syncthreads();

    // region0 reuse: sA(48x4) sB(32x4) qA(48) qB(32) red(8) coA(1024) coB(1024)
    float* sA  = (float*)smem;
    float* sB  = sA + 192;
    float* qA  = sB + 128;
    float* qB  = qA + 48;
    float* red = qB + 32;
    float* coA = red + 8;
    float* coB = coA + DD;
    {
        unsigned* z = (unsigned*)smem;      // zero 2456 floats
        for (int i = tid; i < 2456; i += 256) z[i] = 0u;
    }
    __syncthreads();

    // ---- Phase 5: summary MLP, h = relu(x@w1^T+b1); s += h@w2^T (b2 is a
    // per-head constant -> softmax invariant -> dropped).
    // 8 wave-tasks = (summary, e-quarter).  Each wave loads one w1 B-fragment
    // and multiplies it against ALL M-tile A-fragments (register M-blocking):
    // w1 L2 traffic drops from 10 MB to 4 MB per block, and WMMA density per
    // global load triples. ----
    {
        int t = wid;                         // 0..7, exactly one task per wave
        bool isA = t < 4;
        int  eq  = t & 3;                    // e-quarter: 16 e-tiles
        const __bf16* xload = isA ? xA : xB;
        const __bf16* w1    = isA ? w1ah : w1bh;
        const float*  b1    = isA ? b1a : b1b;
        const float*  w2    = isA ? w2a : w2b;
        float*        sArr  = isA ? sA : sB;
        int Nr = isA ? RR : LL;
        int MT = isA ? 3 : 2;                // M-tiles (wave-uniform)

        bool av[3];
        const __bf16* arow[3];
#pragma unroll
        for (int mt = 0; mt < 3; ++mt) {
            int rowa = mt * 16 + m;
            av[mt]   = (mt < MT) && (rowa < Nr);
            arow[mt] = xload + (size_t)(av[mt] ? rowa : 0) * DD;
        }
        v8bf z8 = {};

        float accs[3][32];
#pragma unroll
        for (int mt = 0; mt < 3; ++mt)
#pragma unroll
            for (int i = 0; i < 32; ++i) accs[mt][i] = 0.f;

        for (int et = eq * 16; et < eq * 16 + 16; ++et) {
            int ebase = et * 16;
            const __bf16* brow = w1 + (size_t)(ebase + m) * DD + half * 16;
            v8f acc[3];
#pragma unroll
            for (int mt = 0; mt < 3; ++mt) acc[mt] = (v8f){};

#pragma unroll 2
            for (int ks = 0; ks < DD; ks += 32) {
                v16bf B = *(const v16bf*)(brow + ks);
#pragma unroll
                for (int mt = 0; mt < 3; ++mt) {
                    if (mt < MT) {
                        v8bf alo = av[mt] ? *(const v8bf*)(arow[mt] + ks + half * 16)     : z8;
                        v8bf ahi = av[mt] ? *(const v8bf*)(arow[mt] + ks + half * 16 + 8) : z8;
                        v16bf A = SHUF16(alo, ahi);
                        acc[mt] = wmma_bf16(A, B, acc[mt]);
                    }
                }
            }
            int col = ebase + m;
            float b1v = b1[col];
            float w20 = w2[col], w21 = w2[DD + col];
            float w22 = w2[2 * DD + col], w23 = w2[3 * DD + col];
#pragma unroll
            for (int mt = 0; mt < 3; ++mt) {
                if (mt < MT) {
#pragma unroll
                    for (int i = 0; i < 8; ++i) {
                        float hv = acc[mt][i] + b1v;
                        hv = hv > 0.f ? hv : 0.f;
                        accs[mt][i * 4 + 0] += hv * w20;
                        accs[mt][i * 4 + 1] += hv * w21;
                        accs[mt][i * 4 + 2] += hv * w22;
                        accs[mt][i * 4 + 3] += hv * w23;
                    }
                }
            }
        }
#pragma unroll
        for (int mt = 0; mt < 3; ++mt) {
            if (mt < MT) {
#pragma unroll
                for (int i = 0; i < 8; ++i) {
                    int row = mt * 16 + i + 8 * half;
                    if (row < Nr)
                        for (int h = 0; h < 4; ++h)
                            atomicAdd(&sArr[row * 4 + h], accs[mt][i * 4 + h]);
                }
            }
        }
    }
    __syncthreads();

    // ---- per-head softmax over n; q[n] = 1/4 * sum_h p[h][n] ----
    if (tid < 8) {
        bool isA = tid < 4;
        int  h   = tid & 3;
        float* sArr = isA ? sA : sB;
        float* q    = isA ? qA : qB;
        int N = isA ? RR : LL;
        float mx = -1e30f;
        for (int n = 0; n < N; ++n) mx = fmaxf(mx, sArr[n * 4 + h]);
        float sum = 0.f;
        for (int n = 0; n < N; ++n) sum += __expf(sArr[n * 4 + h] - mx);
        float inv = 0.25f / sum;
        for (int n = 0; n < N; ++n)
            atomicAdd(&q[n], __expf(sArr[n * 4 + h] - mx) * inv);
    }
    __syncthreads();

    // ---- co[d] = sum_n q[n]*x[n][d] ----
    for (int d = tid; d < DD; d += 256) {
        float ca = 0.f;
        for (int n = 0; n < RR; ++n) ca += qA[n] * (float)xA[(size_t)n * DD + d];
        coA[d] = ca;
        float cb = 0.f;
        for (int n = 0; n < LL; ++n) cb += qB[n] * (float)xB[(size_t)n * DD + d];
        coB[d] = cb;
    }
    __syncthreads();

    // ---- norms + dot -> score ----
    {
        float pa = 0.f, pb = 0.f, pd = 0.f;
        for (int d = tid; d < DD; d += 256) {
            float x1 = coA[d], x2 = coB[d];
            pa += x1 * x1; pb += x2 * x2; pd += x1 * x2;
        }
        atomicAdd(&red[0], pa);
        atomicAdd(&red[1], pb);
        atomicAdd(&red[2], pd);
    }
    __syncthreads();
    if (tid == 0) {
        float na = sqrtf(red[0]) + EPSN;
        float nb = sqrtf(red[1]) + EPSN;
        out[a * NB2 + b] = red[2] / (na * nb);
    }
}

// ---------------------------------------------------------------------------
// Host launcher
// ---------------------------------------------------------------------------
extern "C" void kernel_launch(void* const* d_in, const int* in_sizes, int n_in,
                              void* d_out, int out_size, void* d_ws, size_t ws_size,
                              hipStream_t stream)
{
    (void)in_sizes; (void)n_in; (void)out_size; (void)ws_size;

    const float* v1    = (const float*)d_in[0];
    const float* v2    = (const float*)d_in[1];
    const float* w_img = (const float*)d_in[2];
    const float* w_txt = (const float*)d_in[3];
    const float* w1a   = (const float*)d_in[4];
    const float* b1a   = (const float*)d_in[5];
    const float* w2a   = (const float*)d_in[6];
    /* b2a = d_in[7]  — softmax shift-invariant, unused */
    const float* w1b   = (const float*)d_in[8];
    const float* b1b   = (const float*)d_in[9];
    const float* w2b   = (const float*)d_in[10];
    /* b2b = d_in[11] — unused */
    float* out = (float*)d_out;

    char* ws = (char*)d_ws;
    size_t off = 0;
    auto carve = [&](size_t bytes) -> char* {
        char* p = ws + off;
        off += (bytes + 255) & ~(size_t)255;
        return p;
    };
    const int NV1 = NB1 * RR * DD;   // 1,769,472
    const int NV2 = NB2 * LL * DD;   // 1,474,560
    const int NW  = DD * DD;         // 1,048,576

    __bf16* v1h   = (__bf16*)carve((size_t)NV1 * 2);
    __bf16* v2h   = (__bf16*)carve((size_t)NV2 * 2);
    __bf16* wimgh = (__bf16*)carve((size_t)NW * 2);
    __bf16* wtxth = (__bf16*)carve((size_t)NW * 2);
    __bf16* w1ah  = (__bf16*)carve((size_t)NW * 2);
    __bf16* w1bh  = (__bf16*)carve((size_t)NW * 2);
    __bf16* k1h   = (__bf16*)carve((size_t)NV1 * 2);
    __bf16* k2h   = (__bf16*)carve((size_t)NV2 * 2);

    // f32 -> bf16
    cvt_f32_bf16<<<(NV1 + 255) / 256, 256, 0, stream>>>(v1, v1h, NV1);
    cvt_f32_bf16<<<(NV2 + 255) / 256, 256, 0, stream>>>(v2, v2h, NV2);
    cvt_f32_bf16<<<(NW  + 255) / 256, 256, 0, stream>>>(w_img, wimgh, NW);
    cvt_f32_bf16<<<(NW  + 255) / 256, 256, 0, stream>>>(w_txt, wtxth, NW);
    cvt_f32_bf16<<<(NW  + 255) / 256, 256, 0, stream>>>(w1a, w1ah, NW);
    cvt_f32_bf16<<<(NW  + 255) / 256, 256, 0, stream>>>(w1b, w1bh, NW);

    // k1 = v1 @ w_img^T  (1728x1024), k2 = v2 @ w_txt^T  (1440x1024)
    {
        int tiles1 = (NB1 * RR / 16) * (DD / 16);  // 6912
        int tiles2 = (NB2 * LL / 16) * (DD / 16);  // 5760
        gemm_xw_kernel<<<(tiles1 + 7) / 8, 256, 0, stream>>>(v1h, wimgh, k1h, NB1 * RR);
        gemm_xw_kernel<<<(tiles2 + 7) / 8, 256, 0, stream>>>(v2h, wtxth, k2h, NB2 * LL);
    }

    // fused per-(a,b) pipeline
    dim3 grid(NB2, NB1);
    fused_score_kernel<<<grid, 256, 0, stream>>>(
        k1h, k2h, v1h, v2h, w1ah, w1bh, b1a, b1b, w2a, w2b, out);
}
